// IPT_dehazenet_79680233276284
// MI455X (gfx1250) — matrix operand order repaired and checked by reading, not compiled, not run on hardware
//
#include <hip/hip_runtime.h>

// ---------------------------------------------------------------------------
// SMSAB block for MI455X (gfx1250, wave32, WMMA bf16 16x16x32).
// Dims: B=2, C=172 (pad 192), M=688 (pad 704), H=W=256, NTOK=131072.
// ---------------------------------------------------------------------------

#define CVAL   172
#define CP     192
#define MVAL   688
#define MP     704
#define HWPIX  65536
#define NTOK   131072   // B * HWPIX
#define TOKB   65536    // tokens per batch

typedef __attribute__((ext_vector_type(16))) __bf16 v16bf;
typedef __attribute__((ext_vector_type(8)))  float  v8f;

union Frag { unsigned u[8]; v16bf v; };

static __device__ __forceinline__ float bf2f(unsigned short h) {
    return __uint_as_float(((unsigned)h) << 16);
}
static __device__ __forceinline__ unsigned short f2bf(float f) {
    unsigned u = __float_as_uint(f);
    return (unsigned short)((u + 0x7FFFu + ((u >> 16) & 1u)) >> 16);  // RNE
}
static __device__ __forceinline__ float gelu_f(float x) {
    return 0.5f * x * (1.0f + erff(x * 0.70710678118654752f));
}
static __device__ __forceinline__ v8f wmma_bf16(const Frag& a, const Frag& b, v8f c) {
    return __builtin_amdgcn_wmma_f32_16x16x32_bf16(false, a.v, false, b.v,
                                                   (short)0, c, false, false);
}
static __device__ __forceinline__ float wave_sum(float s) {
    #pragma unroll
    for (int o = 16; o > 0; o >>= 1) s += __shfl_down(s, o, 32);
    return __shfl(s, 0, 32);
}
static __device__ __forceinline__ void load_frag(Frag& f, const uint4* p0, const uint4* p1) {
    uint4 lo = *p0, hi = *p1;
    f.u[0] = lo.x; f.u[1] = lo.y; f.u[2] = lo.z; f.u[3] = lo.w;
    f.u[4] = hi.x; f.u[5] = hi.y; f.u[6] = hi.z; f.u[7] = hi.w;
}

// --------------------------- NCHW -> token-major ---------------------------
__global__ void transpose_in_kernel(const float* __restrict__ x, float* __restrict__ y0) {
    __shared__ float t[32][33];
    int nb = blockIdx.x * 32, cb = blockIdx.y * 32;
    #pragma unroll
    for (int j = 0; j < 4; ++j) {
        int cl = threadIdx.y + 8 * j;
        int c  = cb + cl;
        int n  = nb + threadIdx.x;
        float v = 0.f;
        if (c < CVAL) {
            int b = n >> 16, pix = n & 65535;
            v = x[((size_t)(b * CVAL + c)) * HWPIX + pix];   // contiguous over tx
        }
        t[cl][threadIdx.x] = v;
    }
    __syncthreads();
    #pragma unroll
    for (int j = 0; j < 4; ++j) {
        int nl = threadIdx.y + 8 * j;
        int cl = threadIdx.x;
        y0[(size_t)(nb + nl) * CP + cb + cl] = t[cl][nl];    // contiguous over tx
    }
}

// --------------------------- token-major -> NCHW ---------------------------
__global__ void transpose_out_kernel(const float* __restrict__ y, float* __restrict__ out) {
    __shared__ float t[32][33];
    int nb = blockIdx.x * 32, cb = blockIdx.y * 32;
    #pragma unroll
    for (int j = 0; j < 4; ++j) {
        int nl = threadIdx.y + 8 * j;
        int c  = cb + threadIdx.x;
        t[nl][threadIdx.x] = y[(size_t)(nb + nl) * CP + c];  // contiguous over tx
    }
    __syncthreads();
    #pragma unroll
    for (int j = 0; j < 4; ++j) {
        int n = nb + threadIdx.x;
        int c = cb + threadIdx.y + 8 * j;
        if (c < CVAL) {
            int b = n >> 16, pix = n & 65535;
            out[((size_t)(b * CVAL + c)) * HWPIX + pix] = t[threadIdx.x][threadIdx.y + 8 * j];
        }
    }
}

// ---------------------- LayerNorm (one wave per token) ----------------------
__global__ void ln_kernel(const float* __restrict__ in, const float* __restrict__ g,
                          const float* __restrict__ bta, unsigned short* __restrict__ out) {
    int wave = threadIdx.x >> 5, lane = threadIdx.x & 31;
    size_t base = ((size_t)blockIdx.x * 8 + wave) * CP;
    float s = 0.f;
    for (int c = lane; c < CVAL; c += 32) s += in[base + c];
    float mean = wave_sum(s) * (1.0f / CVAL);
    float v = 0.f;
    for (int c = lane; c < CVAL; c += 32) { float d = in[base + c] - mean; v += d * d; }
    float rstd = rsqrtf(wave_sum(v) * (1.0f / CVAL) + 1e-5f);
    for (int c = lane; c < CP; c += 32) {
        float o = 0.f;
        if (c < CVAL) o = (in[base + c] - mean) * rstd * g[c] + bta[c];
        out[base + c] = f2bf(o);
    }
}

// --------- pack W[O][I] -> bf16 K-tiled B fragments (lane-contiguous) -------
// For pair p = i/2: wp[ ((p>>4)*OP + o)*16 + (p&15) ] = {bf16 W[o][2p], bf16 W[o][2p+1]}
// => a lane's 8 fragment dwords are 2 contiguous uint4 (32B), wave covers 1KB.
__global__ void pack_w_kernel(const float* __restrict__ W, unsigned* __restrict__ wp,
                              int O, int I, int OPd, int IPd) {
    int id = blockIdx.x * 256 + threadIdx.x;
    int tot = (IPd >> 1) * OPd;
    if (id >= tot) return;
    int o = id % OPd, p = id / OPd, i0 = 2 * p;
    float a = (o < O && i0 < I)     ? W[(size_t)o * I + i0]     : 0.f;
    float b = (o < O && i0 + 1 < I) ? W[(size_t)o * I + i0 + 1] : 0.f;
    wp[((size_t)(p >> 4) * OPd + o) * 16 + (p & 15)] =
        (unsigned)f2bf(a) | ((unsigned)f2bf(b) << 16);
}

// ------------------- generic fused WMMA GEMM (bf16 -> f32) ------------------
// out[n,o] = act( sum_i A[n,i]*W[o,i] + bias[o] + resF[n,o] + resB[n,o] )
// grid: (tokens/128, OPD/64); block 256 (8 waves, 16 token rows per wave).
template <int IPD, int OPD>
__global__ __launch_bounds__(256)
void gemm_bf16_kernel(const unsigned short* __restrict__ A, const unsigned* __restrict__ Wp,
                      int Ovalid, const float* __restrict__ bias,
                      const float* __restrict__ resF, const unsigned short* __restrict__ resB,
                      float* __restrict__ outF, unsigned short* __restrict__ outB, int act) {
    int wave = threadIdx.x >> 5, lane = threadIdx.x & 31;
    int hi = lane >> 4, cl = lane & 15;
    size_t tokBase = (size_t)blockIdx.x * 128 + wave * 16;
    int nb = blockIdx.y * 64;
    v8f acc[4];
    #pragma unroll
    for (int j = 0; j < 4; ++j) acc[j] = v8f{};

    const uint4* a4 = (const uint4*)((const unsigned*)A +
                                     (tokBase + cl) * (size_t)(IPD / 2) + hi * 8);
    const unsigned* wrow = Wp + ((size_t)nb + cl) * 16 + hi * 8;

    auto kstep = [&]() {
        Frag a;
        load_frag(a, a4, a4 + 1);
        a4 += 4;                                    // +16 dwords (next K tile)
        #pragma unroll
        for (int j = 0; j < 4; ++j) {
            Frag b;
            load_frag(b, (const uint4*)(wrow + j * 256), (const uint4*)(wrow + j * 256) + 1);
            acc[j] = wmma_bf16(a, b, acc[j]);
        }
        wrow += (size_t)OPD * 16;                   // next K tile of B
    };
    if (IPD == 192) {                               // full unroll: 24 straight-line WMMAs
        #pragma unroll
        for (int kt = 0; kt < 6; ++kt) kstep();
    } else {
        #pragma unroll 2
        for (int kt = 0; kt < IPD / 32; ++kt) kstep();
    }

    #pragma unroll
    for (int j = 0; j < 4; ++j) {
        int col = nb + j * 16 + cl;
        float bi = (bias && col < Ovalid) ? bias[col] : 0.f;
        #pragma unroll
        for (int r = 0; r < 8; ++r) {
            size_t row = tokBase + r + hi * 8;
            size_t idx = row * (size_t)OPD + col;
            float v = acc[j][r] + bi;
            if (resF) v += resF[idx];
            if (resB) v += bf2f(resB[idx]);
            if (act)  v = gelu_f(v);
            if (outF) outF[idx] = v;
            if (outB) outB[idx] = f2bf(v);
        }
    }
}

// ----------------- fused q/k/v projection: A read once, 12 WMMAs/K-step -----
// Wp = [wq|wk|wv] contiguous (each 96*192 dwords); Out = [qb|kb|vb] contiguous.
__global__ __launch_bounds__(256)
void gemm_qkv_kernel(const unsigned short* __restrict__ A, const unsigned* __restrict__ Wp,
                     unsigned short* __restrict__ Out) {
    int wave = threadIdx.x >> 5, lane = threadIdx.x & 31;
    int hi = lane >> 4, cl = lane & 15;
    size_t tokBase = (size_t)blockIdx.x * 128 + wave * 16;
    int nb = blockIdx.y * 64;
    v8f acc[3][4];
    #pragma unroll
    for (int m = 0; m < 3; ++m)
        #pragma unroll
        for (int j = 0; j < 4; ++j) acc[m][j] = v8f{};

    const uint4* a4 = (const uint4*)((const unsigned*)A +
                                     (tokBase + cl) * (size_t)(CP / 2) + hi * 8);
    const unsigned* wrow = Wp + ((size_t)nb + cl) * 16 + hi * 8;

    #pragma unroll 2
    for (int kt = 0; kt < 6; ++kt) {
        Frag a;
        load_frag(a, a4, a4 + 1);
        a4 += 4;
        #pragma unroll
        for (int m = 0; m < 3; ++m) {
            const unsigned* wm = wrow + (size_t)m * ((CP / 2) * CP);
            #pragma unroll
            for (int j = 0; j < 4; ++j) {
                Frag b;
                load_frag(b, (const uint4*)(wm + j * 256), (const uint4*)(wm + j * 256) + 1);
                acc[m][j] = wmma_bf16(a, b, acc[m][j]);
            }
        }
        wrow += (size_t)CP * 16;
    }

    #pragma unroll
    for (int m = 0; m < 3; ++m)
        #pragma unroll
        for (int j = 0; j < 4; ++j) {
            int col = nb + j * 16 + cl;
            #pragma unroll
            for (int r = 0; r < 8; ++r) {
                size_t row = tokBase + r + hi * 8;
                Out[(size_t)m * NTOK * CP + row * CP + col] = f2bf(acc[m][j][r]);
            }
        }
}

// ------------- per-channel sum of squares over tokens (for l2norm) ----------
__global__ void rowsq_kernel(const unsigned short* __restrict__ X, float* __restrict__ out) {
    int b = blockIdx.y, c = threadIdx.x;
    size_t n0 = (size_t)b * TOKB + (size_t)blockIdx.x * 2048;
    float s = 0.f;
    for (int t = 0; t < 2048; ++t) { float v = bf2f(X[(n0 + t) * CP + c]); s += v * v; }
    atomicAdd(&out[b * CP + c], s);
}

// ---------------- Gram: G[b,d,e] = sum_n k[n,d]*q[n,e] via WMMA -------------
// LDS-staged token-axis transpose (pairs along K=tokens). 11 waves = 11 e-tiles.
// 352 threads = 2*176: division-free staging with loop-invariant strides.
__global__ __launch_bounds__(352)
void gram_kernel(const unsigned short* __restrict__ K_, const unsigned short* __restrict__ Q_,
                 float* __restrict__ G) {
    __shared__ unsigned short qld[32 * 176];  // [tokpair][e][2]
    __shared__ unsigned short kld[32 * 16];   // [tokpair][d][2]
    int b = blockIdx.z, dtile = blockIdx.y;
    size_t n0 = (size_t)b * TOKB + (size_t)blockIdx.x * 2048;
    int wave = threadIdx.x >> 5, lane = threadIdx.x & 31;
    int hi = lane >> 4, cl = lane & 15;
    int et = threadIdx.x % 176;               // staged channel (hoisted, once)
    int t2 = threadIdx.x / 176;               // token parity (0/1)
    v8f acc = {};
    const unsigned* kb32 = (const unsigned*)kld + cl + hi * 128;       // + i*16
    const unsigned* qb32 = (const unsigned*)qld + (wave * 16 + cl) + hi * 8 * 176;  // + i*176
    unsigned short* qdst = qld + et * 2 + t2;                          // + s*352
    for (int it = 0; it < 64; ++it) {
        size_t nbt = n0 + it * 32;
        const unsigned short* qs = Q_ + (nbt + t2) * CP + et;          // + s*2*CP
        #pragma unroll
        for (int s = 0; s < 16; ++s)
            qdst[s * 352] = qs[(size_t)s * 2 * CP];
        #pragma unroll
        for (int j = threadIdx.x; j < 512; j += 352) {                 // k tile (shift/mask)
            int tok = j >> 4, d = j & 15;
            kld[(tok >> 1) * 32 + d * 2 + (tok & 1)] = K_[(nbt + tok) * CP + dtile * 16 + d];
        }
        __syncthreads();
        Frag a, bf;
        #pragma unroll
        for (int i = 0; i < 8; ++i)
            a.u[i] = kb32[i * 16];
        #pragma unroll
        for (int i = 0; i < 8; ++i)
            bf.u[i] = qb32[i * 176];
        acc = wmma_bf16(a, bf, acc);
        __syncthreads();
    }
    #pragma unroll
    for (int r = 0; r < 8; ++r) {
        int row = dtile * 16 + r + hi * 8;
        int col = wave * 16 + cl;
        atomicAdd(&G[((size_t)b * CP + row) * CP + col], acc[r]);
    }
}

// ------- softmax over e of G/(||k_d||*||q_e||)*rescale + bf16 B-packing -----
__global__ void softmax_pack_kernel(const float* __restrict__ G, const float* __restrict__ kn,
                                    const float* __restrict__ qn, const float* __restrict__ rescale,
                                    unsigned* __restrict__ ap) {
    __shared__ float red[256];
    __shared__ float p[CP];
    int b = blockIdx.y, d = blockIdx.x, e = threadIdx.x;
    float scale = rescale[0];
    bool valid = (d < CVAL) && (e < CVAL);
    float sv = 0.f;
    if (valid) {
        float kd = fmaxf(sqrtf(kn[b * CP + d]), 1e-12f);
        float qe = fmaxf(sqrtf(qn[b * CP + e]), 1e-12f);
        sv = G[((size_t)b * CP + d) * CP + e] / (kd * qe) * scale;
    }
    red[e] = valid ? sv : -3.4e38f;
    __syncthreads();
    for (int s = 128; s > 0; s >>= 1) { if (e < s) red[e] = fmaxf(red[e], red[e + s]); __syncthreads(); }
    float mx = red[0];
    __syncthreads();
    float ex = valid ? __expf(sv - mx) : 0.f;
    red[e] = ex;
    __syncthreads();
    for (int s = 128; s > 0; s >>= 1) { if (e < s) red[e] += red[e + s]; __syncthreads(); }
    float sm = red[0];
    float pr = (valid && sm > 0.f) ? ex / sm : 0.f;
    if (e < CP) p[e] = pr;
    __syncthreads();
    if (e < CP && !(e & 1)) {
        int pp = e >> 1;  // pair index; same K-tiled fragment layout as pack_w
        ap[((size_t)b * (CP / 2) * CP) + ((size_t)(pp >> 4) * CP + d) * 16 + (pp & 15)] =
            (unsigned)f2bf(p[e]) | ((unsigned)f2bf(p[e + 1]) << 16);
    }
}

// -------- depthwise 3x3 conv, bf16 channel-pair vectorized, LDS weights -----
__global__ void dwconv_kernel(const unsigned short* __restrict__ in, const float* __restrict__ w,
                              unsigned short* __restrict__ out, int S, int Cvalid, int act) {
    __shared__ float wl[MP * 9];
    for (int i = threadIdx.x; i < S * 9; i += 256) wl[i] = w[i];
    __syncthreads();
    int n = blockIdx.x;
    int b = n >> 16, y = (n >> 8) & 255, x = n & 255;
    const unsigned* in32 = (const unsigned*)in;
    unsigned* out32 = (unsigned*)out;
    int S2 = S >> 1;
    for (int cp = threadIdx.x; cp < S2; cp += 256) {
        int c = cp * 2;
        float a0 = 0.f, a1 = 0.f;
        if (c < Cvalid) {   // Cvalid even: pairs never straddle the valid edge
            #pragma unroll
            for (int dy = -1; dy <= 1; ++dy) {
                int yy = y + dy; if (yy < 0 || yy > 255) continue;
                #pragma unroll
                for (int dx = -1; dx <= 1; ++dx) {
                    int xx = x + dx; if (xx < 0 || xx > 255) continue;
                    size_t nn = (size_t)((b << 16) | (yy << 8) | xx);
                    unsigned pk = in32[nn * S2 + cp];
                    float wt0 = wl[c * 9 + (dy + 1) * 3 + (dx + 1)];
                    float wt1 = wl[(c + 1) * 9 + (dy + 1) * 3 + (dx + 1)];
                    a0 += wt0 * bf2f((unsigned short)(pk & 0xFFFF));
                    a1 += wt1 * bf2f((unsigned short)(pk >> 16));
                }
            }
        }
        if (act) { a0 = gelu_f(a0); a1 = gelu_f(a1); }
        out32[(size_t)n * S2 + cp] = (unsigned)f2bf(a0) | ((unsigned)f2bf(a1) << 16);
    }
}

__global__ void zero_f32_kernel(float* p, int n) {
    int i = blockIdx.x * 256 + threadIdx.x;
    if (i < n) p[i] = 0.f;
}

// ---------------------------------------------------------------------------
extern "C" void kernel_launch(void* const* d_in, const int* in_sizes, int n_in,
                              void* d_out, int out_size, void* d_ws, size_t ws_size,
                              hipStream_t stream) {
    const float* x      = (const float*)d_in[0];
    const float* ln1_g  = (const float*)d_in[1];
    const float* ln1_b  = (const float*)d_in[2];
    const float* Wq     = (const float*)d_in[3];
    const float* Wk     = (const float*)d_in[4];
    const float* Wv     = (const float*)d_in[5];
    const float* rescale= (const float*)d_in[6];
    const float* proj_w = (const float*)d_in[7];
    const float* proj_b = (const float*)d_in[8];
    const float* pos_w1 = (const float*)d_in[9];
    const float* pos_w2 = (const float*)d_in[10];
    const float* ln2_g  = (const float*)d_in[11];
    const float* ln2_b  = (const float*)d_in[12];
    const float* ff_w1  = (const float*)d_in[13];
    const float* ff_dw  = (const float*)d_in[14];
    const float* ff_w2  = (const float*)d_in[15];

    char* ws = (char*)d_ws;
    const size_t SZ_F32 = (size_t)NTOK * CP * 4;   // 100.7 MB
    const size_t SZ_BF  = (size_t)NTOK * CP * 2;   //  50.3 MB
    const size_t SZ_BFM = (size_t)NTOK * MP * 2;   // 184.5 MB
    size_t o = 0;
    float*          y0  = (float*)(ws + o);          o += SZ_F32;  // token-major x / final y2
    unsigned short* a   = (unsigned short*)(ws + o); o += SZ_BF;   // LN output (reused for LN2)
    unsigned short* qb  = (unsigned short*)(ws + o); o += SZ_BF;   // qb|kb|vb contiguous
    unsigned short* kb  = (unsigned short*)(ws + o); o += SZ_BF;
    unsigned short* vb  = (unsigned short*)(ws + o); o += SZ_BF;
    unsigned short* xo  = (unsigned short*)(ws + o); o += SZ_BF;
    unsigned short* pe1 = (unsigned short*)(ws + o); o += SZ_BF;
    unsigned short* pe2 = (unsigned short*)(ws + o); o += SZ_BF;
    float*          y1  = (float*)(ws + o);          o += SZ_F32;
    unsigned short* h1  = (unsigned short*)(ws + o); o += SZ_BFM;
    unsigned short* h2  = (unsigned short*)(ws + o); o += SZ_BFM;
    unsigned* wqp  = (unsigned*)(ws + o); o += (size_t)(CP/2) * CP * 4;  // wq|wk|wv contiguous
    unsigned* wkp  = (unsigned*)(ws + o); o += (size_t)(CP/2) * CP * 4;
    unsigned* wvp  = (unsigned*)(ws + o); o += (size_t)(CP/2) * CP * 4;
    unsigned* wpp  = (unsigned*)(ws + o); o += (size_t)(CP/2) * CP * 4;
    unsigned* w1p  = (unsigned*)(ws + o); o += (size_t)(CP/2) * MP * 4;
    unsigned* w2p  = (unsigned*)(ws + o); o += (size_t)(MP/2) * CP * 4;
    unsigned* ap   = (unsigned*)(ws + o); o += (size_t)2 * (CP/2) * CP * 4;
    float*    G    = (float*)(ws + o);    o += (size_t)2 * CP * CP * 4;
    float*    qn   = (float*)(ws + o);    o += (size_t)2 * CP * 4;
    float*    kn   = (float*)(ws + o);    o += (size_t)2 * CP * 4;

    dim3 tb(32, 8), tg(NTOK / 32, CP / 32);
    dim3 gemmC(NTOK / 128, CP / 64);   // (1024, 3)
    dim3 gemmM(NTOK / 128, MP / 64);   // (1024, 11)
    dim3 gemmB(TOKB / 128, CP / 64);   // (512, 3) per-batch

    // 0) weight packing (independent)
    pack_w_kernel<<<((CP/2)*CP + 255) / 256, 256, 0, stream>>>(Wq,     wqp, CVAL, CVAL, CP, CP);
    pack_w_kernel<<<((CP/2)*CP + 255) / 256, 256, 0, stream>>>(Wk,     wkp, CVAL, CVAL, CP, CP);
    pack_w_kernel<<<((CP/2)*CP + 255) / 256, 256, 0, stream>>>(Wv,     wvp, CVAL, CVAL, CP, CP);
    pack_w_kernel<<<((CP/2)*CP + 255) / 256, 256, 0, stream>>>(proj_w, wpp, CVAL, CVAL, CP, CP);
    pack_w_kernel<<<((CP/2)*MP + 255) / 256, 256, 0, stream>>>(ff_w1,  w1p, MVAL, CVAL, MP, CP);
    pack_w_kernel<<<((MP/2)*CP + 255) / 256, 256, 0, stream>>>(ff_w2,  w2p, CVAL, MVAL, CP, MP);
    zero_f32_kernel<<<(2*CP*CP + 2*2*CP + 255) / 256, 256, 0, stream>>>(G, 2*CP*CP + 2*2*CP);

    // 1) NCHW -> token-major, LN1
    transpose_in_kernel<<<tg, tb, 0, stream>>>(x, y0);
    ln_kernel<<<NTOK / 8, 256, 0, stream>>>(y0, ln1_g, ln1_b, a);

    // 2) fused q,k,v projections (A read once, 12 WMMAs per K-step)
    gemm_qkv_kernel<<<gemmC, 256, 0, stream>>>(a, wqp, qb);

    // 3) l2-norm stats + Gram matrix (WMMA over 65536-token K axis) + softmax
    rowsq_kernel<<<dim3(32, 2), CP, 0, stream>>>(qb, qn);
    rowsq_kernel<<<dim3(32, 2), CP, 0, stream>>>(kb, kn);
    gram_kernel<<<dim3(32, 11, 2), 352, 0, stream>>>(kb, qb, G);
    softmax_pack_kernel<<<dim3(CP, 2), 256, 0, stream>>>(G, kn, qn, rescale, ap);

    // 4) positional branch: GELU(dwconv(v)) -> dwconv
    dwconv_kernel<<<NTOK, 256, 0, stream>>>(vb,  pos_w1, pe1, CP, CVAL, 1);
    dwconv_kernel<<<NTOK, 256, 0, stream>>>(pe1, pos_w2, pe2, CP, CVAL, 0);

    // 5) xo = attn @ v (per-batch attention weights), then fused proj+bias+pos+residual
    for (int b = 0; b < 2; ++b) {
        gemm_bf16_kernel<CP, CP><<<gemmB, 256, 0, stream>>>(
            vb + (size_t)b * TOKB * CP, ap + (size_t)b * (CP / 2) * CP,
            CVAL, nullptr, nullptr, nullptr,
            nullptr, xo + (size_t)b * TOKB * CP, 0);
    }
    gemm_bf16_kernel<CP, CP><<<gemmC, 256, 0, stream>>>(xo, wpp, CVAL, proj_b, y0, pe2, y1, nullptr, 0);

    // 6) FFN: LN2 -> GELU(1x1 expand) -> GELU(3x3 dw) -> 1x1 project + residual
    ln_kernel<<<NTOK / 8, 256, 0, stream>>>(y1, ln2_g, ln2_b, a);
    gemm_bf16_kernel<CP, MP><<<gemmM, 256, 0, stream>>>(a, w1p, MVAL, nullptr, nullptr, nullptr, nullptr, h1, 1);
    dwconv_kernel<<<NTOK, 256, 0, stream>>>(h1, ff_dw, h2, MP, MVAL, 1);
    gemm_bf16_kernel<MP, CP><<<gemmC, 256, 0, stream>>>(h2, w2p, CVAL, nullptr, y1, nullptr, y0, nullptr, 0);

    // 7) token-major -> NCHW output
    transpose_out_kernel<<<tg, tb, 0, stream>>>(y0, (float*)d_out);

    (void)in_sizes; (void)n_in; (void)out_size; (void)ws_size;
}
